// ResNet18_Scale_34376918237933
// MI455X (gfx1250) — matrix operand order repaired and checked by reading, not compiled, not run on hardware
//
#include <hip/hip_runtime.h>
#include <math.h>

// CDNA5 wave32 WMMA types: v16h = 8 VGPRs of packed f16 (A/B operands),
// v8f = 8 VGPRs f32 (16x16 C/D accumulator tile).
typedef __attribute__((ext_vector_type(16))) _Float16 v16h;
typedef __attribute__((ext_vector_type(8)))  float    v8f;
// Tensor DMA descriptor groups (CDNA5 ISA)
typedef __attribute__((ext_vector_type(4))) unsigned int v4u;
typedef __attribute__((ext_vector_type(8))) int          v8i;
typedef __attribute__((ext_vector_type(4))) int          v4i;

// Largest K = Cin*KH*KW in this network: layer4 3x3 conv on 512ch = 4608.
#define KMAX 4608

#if __has_builtin(__builtin_amdgcn_tensor_load_to_lds)
#define HAVE_TDM 1
#else
#define HAVE_TDM 0
#endif

#if HAVE_TDM
// Issue a TDM load of a 2D f32 tile (tileH rows x tileW cols) from a row-major
// matrix (row stride rowStride elements) into LDS at ldsAddr. dim0/dim1 are the
// remaining tensor extents measured from the tile start: the TDM zero-fills
// reads beyond them, giving free K/Cout edge handling.
__device__ __forceinline__ void tdm_load_tile_f32(
    unsigned ldsAddr, const float* gptr, int dim0, int dim1,
    int tileW, int tileH, int rowStride)
{
  unsigned long long ga = (unsigned long long)(size_t)gptr;
  v4u g0;
  g0.x = 1u;                                           // count=1 (valid), user mode
  g0.y = ldsAddr;                                      // lds_addr [63:32]
  g0.z = (unsigned)(ga & 0xffffffffu);                 // global_addr[31:0]
  g0.w = (unsigned)((ga >> 32) & 0x1ffffffu) | (2u << 30);  // addr[56:32] | type=2
  v8i g1;
  g1[0] = (int)(2u << 16);                             // data_size=2 -> 4 bytes
  g1[1] = (int)(((unsigned)dim0 & 0xffffu) << 16);     // tensor_dim0[15:0]
  g1[2] = (int)((((unsigned)dim0 >> 16) & 0xffffu) |
                (((unsigned)dim1 & 0xffffu) << 16));   // dim0[31:16] | dim1[15:0]
  g1[3] = (int)((((unsigned)dim1 >> 16) & 0xffffu) |
                ((unsigned)tileW << 16));              // dim1[31:16] | tile_dim0
  g1[4] = (int)((unsigned)tileH);                      // tile_dim1 | tile_dim2=0
  g1[5] = (int)((unsigned)rowStride);                  // tensor_dim0_stride[31:0]
  g1[6] = 0;
  g1[7] = 0;
  v4i z4 = {0, 0, 0, 0};
#if defined(__clang_major__) && (__clang_major__ >= 23)
  v8i z8 = {0, 0, 0, 0, 0, 0, 0, 0};
  __builtin_amdgcn_tensor_load_to_lds(g0, g1, z4, z4, z8, 0);
#else
  __builtin_amdgcn_tensor_load_to_lds(g0, g1, z4, z4, 0);
#endif
}
#endif

// ---------------------------------------------------------------------------
// Implicit-GEMM convolution on v_wmma_f32_16x16x32_f16.
// GEMM view: M = Cout, N = B*Ho*Wo, K = Cin*KH*KW.
// Each wave computes a 32(M) x 16(N) tile: two WMMAs share one B (im2col) tile.
// The 32x32 f32 weight chunk is DMA'd into LDS by the Tensor Data Mover
// (double-buffered, s_wait_tensorcnt-paced) by wave 0 and shared by all 4
// waves. The im2col k -> (ci,kr,kc) decomposition is precomputed per block
// into an LDS table (one int div per k per block).
// ---------------------------------------------------------------------------
__global__ void __launch_bounds__(128)
conv_wmma_kernel(const float* __restrict__ in, const float* __restrict__ wt,
                 float* __restrict__ out,
                 int B, int Cin, int H, int W,
                 int Cout, int Ho, int Wo,
                 int KH, int KW, int stride, int pad)
{
  __shared__ int            kdelta[KMAX];     // (ci*H + kr)*W + kc
  __shared__ unsigned short kpack[KMAX];      // kr<<8 | kc
  __shared__ float          Astage[2][32 * 32];  // double-buffered weight tiles

  const int K      = Cin * KH * KW;
  const int KHW    = KH * KW;
  const int Npix   = B * Ho * Wo;
  const int m32    = (Cout + 31) >> 5;
  const int ntiles = (Npix + 15) >> 4;

  // cooperative im2col table build
  for (int k = threadIdx.x; k < K; k += blockDim.x) {
    int ci  = k / KHW;
    int rem = k - ci * KHW;
    int kr  = rem / KW;
    int kc  = rem - kr * KW;
    kdelta[k] = (ci * H + kr) * W + kc;
    kpack[k]  = (unsigned short)((kr << 8) | kc);
  }
  __syncthreads();

  const int wave  = threadIdx.x >> 5;
  const int mt    = blockIdx.x % m32;             // 32-channel tile (block-shared)
  const int ntRaw = (blockIdx.x / m32) * 4 + wave;
  // No early exit: barriers live in the K loop, so idle tail waves compute a
  // valid dummy tile and just skip the stores.
  const bool waveActive = ntRaw < ntiles;
  const int  nt = waveActive ? ntRaw : (ntiles - 1);

  const int lane = threadIdx.x & 31;
  const int g    = lane >> 4;                    // half-wave group
  const int r16  = lane & 15;

  const int  n      = nt * 16 + r16;             // B column (output pixel)
  const bool nValid = (n < Npix) && waveActive;
  int img = 0, ho = 0, wo = 0;
  {
    int nc  = (n < Npix) ? n : 0;
    img = nc / (Ho * Wo);
    int rem = nc - img * (Ho * Wo);
    ho = rem / Wo;
    wo = rem - ho * Wo;
  }
  const float* inBase = in + (size_t)img * Cin * H * W;
  const int ihb = ho * stride - pad;
  const int iwb = wo * stride - pad;
  const int pixOff = ihb * W + iwb;

  const float* wbase   = wt + (size_t)mt * 32 * K;  // first weight row of tile
  const int    rowsRem = Cout - mt * 32;            // rows left (TDM zero-fills)
  const int    nch     = (K + 31) >> 5;             // number of 32-wide K chunks

  v8f acc0 = {};
  v8f acc1 = {};

#if HAVE_TDM
  if (wave == 0)
    tdm_load_tile_f32((unsigned)(size_t)&Astage[0][0], wbase, K, rowsRem, 32, 32, K);
#endif

  for (int c = 0; c < nch; ++c) {
    const int k0 = c << 5;
#if HAVE_TDM
    if (wave == 0) {
      if (c + 1 < nch) {
        tdm_load_tile_f32((unsigned)(size_t)&Astage[(c + 1) & 1][0],
                          wbase + (size_t)(k0 + 32), K - (k0 + 32), rowsRem,
                          32, 32, K);
        __builtin_amdgcn_s_wait_tensorcnt(1);   // current chunk's DMA done
      } else {
        __builtin_amdgcn_s_wait_tensorcnt(0);   // drain last DMA
      }
    }
    __syncthreads();                            // A tile visible to all waves
#else
    __syncthreads();                            // previous readers done
    for (int t = threadIdx.x; t < 32 * 32; t += 128) {
      int row = t >> 5, col = t & 31;
      Astage[c & 1][t] = (row < rowsRem && (k0 + col) < K)
                             ? wbase[(size_t)row * K + k0 + col] : 0.f;
    }
    __syncthreads();
#endif
    const float* As = &Astage[c & 1][0];

    v16h a0, a1, bb;
    // A 16x32 f16 layout: lanes 0-15 M rows, K-pairs 0..7/16..23; lanes 16-31 +8 K
#pragma unroll
    for (int e = 0; e < 8; ++e) {
      int kk = ((e < 4) ? (e << 1) : (16 + ((e - 4) << 1))) + (g << 3);
      a0[2 * e]     = (_Float16)As[r16 * 32 + kk];
      a0[2 * e + 1] = (_Float16)As[r16 * 32 + kk + 1];
      a1[2 * e]     = (_Float16)As[(16 + r16) * 32 + kk];
      a1[2 * e + 1] = (_Float16)As[(16 + r16) * 32 + kk + 1];
    }
    // B 32x16 f16 layout: lanes 0-15 N=lane, K=0..7/16..23; lanes 16-31 N=lane-16, +8 K
#pragma unroll
    for (int v = 0; v < 8; ++v) {
      int kb = k0 + ((v < 4) ? (v << 1) : (16 + ((v - 4) << 1))) + (g << 3);
#pragma unroll
      for (int h2 = 0; h2 < 2; ++h2) {
        int kk = kb + h2;
        float val = 0.f;
        if (nValid && kk < K) {
          int d  = kdelta[kk];
          int pk = kpack[kk];
          int kr = pk >> 8, kc = pk & 255;
          if ((unsigned)(ihb + kr) < (unsigned)H && (unsigned)(iwb + kc) < (unsigned)W)
            val = inBase[pixOff + d];
        }
        bb[2 * v + h2] = (_Float16)val;
      }
    }
    acc0 = __builtin_amdgcn_wmma_f32_16x16x32_f16(false, a0, false, bb, (short)0, acc0,
                                                  false, false);
    acc1 = __builtin_amdgcn_wmma_f32_16x16x32_f16(false, a1, false, bb, (short)0, acc1,
                                                  false, false);
    __syncthreads();                            // readers done before next DMA lands
  }

  // C/D layout: VGPR i, lanes 0-15 -> M=i, lanes 16-31 -> M=i+8; N = lane%16
  if (nValid) {
#pragma unroll
    for (int i = 0; i < 8; ++i) {
      int m  = i + (g << 3);
      int c0 = mt * 32 + m;
      if (c0 < Cout)
        out[(((size_t)img * Cout + c0) * Ho + ho) * Wo + wo] = acc0[i];
      int c1 = mt * 32 + 16 + m;
      if (c1 < Cout)
        out[(((size_t)img * Cout + c1) * Ho + ho) * Wo + wo] = acc1[i];
    }
  }
}

// ---------------------------------------------------------------------------
// BatchNorm (batch statistics): per-channel mean/var reduction, then fused
// normalize + optional residual add + optional ReLU.
// ---------------------------------------------------------------------------
__global__ void __launch_bounds__(256)
bn_stats_kernel(const float* __restrict__ x, float* __restrict__ mean,
                float* __restrict__ var, int B, int C, int HW)
{
  const int c = blockIdx.x;
  __shared__ float ssum[256];
  __shared__ float ssq[256];
  float s = 0.f, q = 0.f;
  const size_t per = (size_t)B * HW;
  for (size_t i = threadIdx.x; i < per; i += blockDim.x) {
    int b  = (int)(i / HW);
    int hw = (int)(i - (size_t)b * HW);
    float v = x[((size_t)b * C + c) * HW + hw];
    s += v;
    q += v * v;
  }
  ssum[threadIdx.x] = s;
  ssq[threadIdx.x]  = q;
  __syncthreads();
  for (int st = 128; st > 0; st >>= 1) {
    if ((int)threadIdx.x < st) {
      ssum[threadIdx.x] += ssum[threadIdx.x + st];
      ssq[threadIdx.x]  += ssq[threadIdx.x + st];
    }
    __syncthreads();
  }
  if (threadIdx.x == 0) {
    float m = ssum[0] / (float)per;
    mean[c] = m;
    var[c]  = ssq[0] / (float)per - m * m;
  }
}

__global__ void __launch_bounds__(256)
bn_apply_kernel(const float* __restrict__ x, const float* __restrict__ mean,
                const float* __restrict__ var, const float* __restrict__ gamma,
                const float* __restrict__ beta, const float* __restrict__ residual,
                float* __restrict__ y, int C, int HW, size_t total, int relu)
{
  size_t i = (size_t)blockIdx.x * blockDim.x + threadIdx.x;
  if (i >= total) return;
  int c = (int)((i / (size_t)HW) % (size_t)C);
  float v = (x[i] - mean[c]) * rsqrtf(var[c] + 1e-5f) * gamma[c] + beta[c];
  if (residual) v += residual[i];
  if (relu) v = fmaxf(v, 0.f);
  y[i] = v;
}

// ---------------------------------------------------------------------------
// Pooling / pixel-shuffle / elementwise kernels
// ---------------------------------------------------------------------------
__global__ void __launch_bounds__(256)
maxpool3s2_kernel(const float* __restrict__ x, float* __restrict__ y,
                  int BC, int H, int W, int Ho, int Wo)
{
  int idx = blockIdx.x * blockDim.x + threadIdx.x;
  int total = BC * Ho * Wo;
  if (idx >= total) return;
  int bc  = idx / (Ho * Wo);
  int rem = idx - bc * (Ho * Wo);
  int ho = rem / Wo, wo = rem - (rem / Wo) * Wo;
  float m = -3.4e38f;
  for (int r = 0; r < 3; ++r)
    for (int c = 0; c < 3; ++c) {
      int ih = ho * 2 - 1 + r;
      int iw = wo * 2 - 1 + c;
      if ((unsigned)ih < (unsigned)H && (unsigned)iw < (unsigned)W)
        m = fmaxf(m, x[((size_t)bc * H + ih) * W + iw]);
    }
  y[idx] = m;
}

__global__ void __launch_bounds__(256)
pixel_shuffle_kernel(const float* __restrict__ x, float* __restrict__ y,
                     int B, int Cout, int r, int H, int W)
{
  // x: (B, Cout*r*r, H, W) -> y: (B, Cout, H*r, W*r)
  size_t total = (size_t)B * Cout * H * r * W * r;
  size_t i = (size_t)blockIdx.x * blockDim.x + threadIdx.x;
  if (i >= total) return;
  int Wr = W * r, Hr = H * r;
  int ww = (int)(i % Wr);
  size_t t = i / Wr;
  int hh = (int)(t % Hr);
  t /= Hr;
  int c = (int)(t % Cout);
  int b = (int)(t / Cout);
  int h = hh / r, ri = hh - h * r;
  int w = ww / r, rj = ww - w * r;
  int cin = c * r * r + ri * r + rj;
  y[i] = x[(((size_t)b * (Cout * r * r) + cin) * H + h) * W + w];
}

__global__ void __launch_bounds__(256)
add4_kernel(const float* __restrict__ a, const float* __restrict__ b,
            const float* __restrict__ c, const float* __restrict__ d,
            float* __restrict__ y, size_t n)
{
  size_t i = (size_t)blockIdx.x * blockDim.x + threadIdx.x;
  if (i < n) y[i] = a[i] + b[i] + c[i] + d[i];
}

__global__ void __launch_bounds__(256)
sigmoid_kernel(const float* __restrict__ x, float* __restrict__ y, size_t n)
{
  size_t i = (size_t)blockIdx.x * blockDim.x + threadIdx.x;
  if (i < n) y[i] = 1.f / (1.f + expf(-x[i]));
}

__global__ void __launch_bounds__(256)
bias_add_kernel(const float* __restrict__ x, const float* __restrict__ bias,
                float* __restrict__ y, int C, int HW, size_t n)
{
  size_t i = (size_t)blockIdx.x * blockDim.x + threadIdx.x;
  if (i >= n) return;
  int c = (int)((i / (size_t)HW) % (size_t)C);
  y[i] = x[i] + bias[c];
}

// ---------------------------------------------------------------------------
// Hash coding: 3x3/3 VALID avg-pool, top-2 (first-index tie break), encode
// row/col with FEAT_SIZE = 8.
// ---------------------------------------------------------------------------
__global__ void __launch_bounds__(256)
hash_coding_kernel(const float* __restrict__ x, float* __restrict__ code,
                   int B, int C, int H, int W)
{
  int idx = blockIdx.x * blockDim.x + threadIdx.x;
  if (idx >= B * C) return;
  const float* p = x + (size_t)idx * H * W;
  int Hp = (H - 3) / 3 + 1, Wp = (W - 3) / 3 + 1;
  float best0 = -3.4e38f, best1 = -3.4e38f;
  int i0 = 0, i1 = 0;
  for (int ph = 0; ph < Hp; ++ph)
    for (int pw = 0; pw < Wp; ++pw) {
      float s = 0.f;
      for (int r = 0; r < 3; ++r)
        for (int c = 0; c < 3; ++c)
          s += p[(ph * 3 + r) * W + (pw * 3 + c)];
      s *= (1.f / 9.f);
      int fi = ph * Wp + pw;
      if (s > best0) { best1 = best0; i1 = i0; best0 = s; i0 = fi; }
      else if (s > best1) { best1 = s; i1 = fi; }
    }
  float* o = code + (size_t)idx * 4;
  o[0] = (float)(i0 / 8); o[1] = (float)(i0 % 8);
  o[2] = (float)(i1 / 8); o[3] = (float)(i1 % 8);
}

// ---------------------------------------------------------------------------
// K-means (4 clusters, 4-dim points, 25 iters) — one block per batch element.
// ---------------------------------------------------------------------------
__global__ void __launch_bounds__(64)
kmeans4_kernel(const float* __restrict__ X, float* __restrict__ cen, int Npts)
{
  int b = blockIdx.x;
  const float* pts = X + (size_t)b * Npts * 4;
  __shared__ float c[4][4];
  __shared__ float sum[4][4];
  __shared__ float cnt[4];
  if (threadIdx.x < 16) c[threadIdx.x >> 2][threadIdx.x & 3] = pts[threadIdx.x];
  __syncthreads();
  for (int it = 0; it < 25; ++it) {
    if (threadIdx.x < 16) sum[threadIdx.x >> 2][threadIdx.x & 3] = 0.f;
    if (threadIdx.x < 4) cnt[threadIdx.x] = 0.f;
    __syncthreads();
    for (int i = threadIdx.x; i < Npts; i += blockDim.x) {
      const float* p = pts + i * 4;
      float bd = 3.4e38f; int bj = 0;
      for (int j = 0; j < 4; ++j) {
        float d = 0.f;
        for (int k = 0; k < 4; ++k) { float t = p[k] - c[j][k]; d += t * t; }
        if (d < bd) { bd = d; bj = j; }   // first-min tie break, matches argmin
      }
      atomicAdd(&cnt[bj], 1.f);
      for (int k = 0; k < 4; ++k) atomicAdd(&sum[bj][k], p[k]);
    }
    __syncthreads();
    if (threadIdx.x < 4) {
      int j = threadIdx.x;
      if (cnt[j] > 0.f)
        for (int k = 0; k < 4; ++k) c[j][k] = sum[j][k] / fmaxf(cnt[j], 1.f);
    }
    __syncthreads();
  }
  if (threadIdx.x < 16) cen[(size_t)b * 16 + threadIdx.x] = c[threadIdx.x >> 2][threadIdx.x & 3];
}

__global__ void __launch_bounds__(64)
cluster_means_kernel(const float* __restrict__ X, const float* __restrict__ cenIn,
                     float* __restrict__ outp, int Npts)
{
  int b = blockIdx.x;
  const float* pts = X + (size_t)b * Npts * 4;
  const float* ce  = cenIn + (size_t)b * 16;
  __shared__ float cn[4][4];
  __shared__ float sum[4][4];
  __shared__ float cnt[4];
  if (threadIdx.x < 4) {
    float nr = 0.f;
    for (int k = 0; k < 4; ++k) nr += ce[threadIdx.x * 4 + k] * ce[threadIdx.x * 4 + k];
    nr = fmaxf(sqrtf(nr), 1e-8f);
    for (int k = 0; k < 4; ++k) cn[threadIdx.x][k] = ce[threadIdx.x * 4 + k] / nr;
    cnt[threadIdx.x] = 0.f;
  }
  if (threadIdx.x < 16) sum[threadIdx.x >> 2][threadIdx.x & 3] = 0.f;
  __syncthreads();
  for (int i = threadIdx.x; i < Npts; i += blockDim.x) {
    const float* p = pts + i * 4;
    float nr = 0.f;
    for (int k = 0; k < 4; ++k) nr += p[k] * p[k];
    nr = fmaxf(sqrtf(nr), 1e-8f);
    float bs = -3.4e38f; int bj = 0;
    for (int j = 0; j < 4; ++j) {
      float s = 0.f;
      for (int k = 0; k < 4; ++k) s += (p[k] / nr) * cn[j][k];
      if (s > bs) { bs = s; bj = j; }    // first-max tie break, matches argmax
    }
    atomicAdd(&cnt[bj], 1.f);
    for (int k = 0; k < 4; ++k) atomicAdd(&sum[bj][k], p[k]);
  }
  __syncthreads();
  if (threadIdx.x < 4) {
    int j = threadIdx.x;
    for (int k = 0; k < 4; ++k)
      outp[(size_t)b * 16 + j * 4 + k] =
          (cnt[j] > 0.f) ? sum[j][k] / fmaxf(cnt[j], 1.f) : ce[j * 4 + k];
  }
}

// ---------------------------------------------------------------------------
// Final reductions + FC
// ---------------------------------------------------------------------------
__global__ void __launch_bounds__(256)
mean_c_kernel(const float* __restrict__ x, float* __restrict__ y, int B, int C, int HW)
{
  int idx = blockIdx.x * blockDim.x + threadIdx.x;
  if (idx >= B * HW) return;
  int b = idx / HW, hw = idx - b * HW;
  float s = 0.f;
  for (int c = 0; c < C; ++c) s += x[((size_t)b * C + c) * HW + hw];
  y[idx] = s / (float)C;
}

__global__ void __launch_bounds__(128)
mean_bc_kernel(const float* __restrict__ x, float* __restrict__ y, int B, int C, int HW)
{
  int hw = blockIdx.x * blockDim.x + threadIdx.x;
  if (hw >= HW) return;
  float s = 0.f;
  for (int i = 0; i < B * C; ++i) s += x[(size_t)i * HW + hw];
  y[hw] = s / (float)(B * C);
}

__global__ void __launch_bounds__(256)
xf_kernel(const float* __restrict__ xmean, const float* __restrict__ gmean,
          float* __restrict__ xf, int B, int HW)
{
  int idx = blockIdx.x * blockDim.x + threadIdx.x;
  if (idx >= B * HW) return;
  xf[idx] = xmean[idx] + gmean[idx % HW];
}

__global__ void __launch_bounds__(128)
fc_kernel(const float* __restrict__ xf, const float* __restrict__ w,
          const float* __restrict__ bias, float* __restrict__ out,
          int B, int Kdim, int Nout)
{
  int idx = blockIdx.x * blockDim.x + threadIdx.x;
  if (idx >= B * Nout) return;
  int b = idx / Nout, j = idx - b * Nout;
  float s = bias[j];
  for (int k = 0; k < Kdim; ++k) s += xf[b * Kdim + k] * w[j * Kdim + k];
  out[idx] = s;
}

// ---------------------------------------------------------------------------
// Host driver
// ---------------------------------------------------------------------------
extern "C" void kernel_launch(void* const* d_in, const int* in_sizes, int n_in,
                              void* d_out, int out_size, void* d_ws, size_t ws_size,
                              hipStream_t stream)
{
  (void)in_sizes; (void)n_in; (void)out_size; (void)ws_size;
  int ii = 0;
  auto nextf = [&]() { return (const float*)d_in[ii++]; };

  const float* X      = nextf();           // (16,3,224,224)
  const float* stem_w = nextf();
  const float* stem_g = nextf();
  const float* stem_b = nextf();

  struct Blk { const float *w1,*g1,*b1,*w2,*g2,*b2,*wd,*gd,*bd; };
  Blk L[8];
  int bi = 0;
  for (int layer = 0; layer < 4; ++layer)
    for (int j = 0; j < 2; ++j) {
      Blk k;
      k.w1 = nextf(); k.g1 = nextf(); k.b1 = nextf();
      k.w2 = nextf(); k.g2 = nextf(); k.b2 = nextf();
      if (layer > 0 && j == 0) { k.wd = nextf(); k.gd = nextf(); k.bd = nextf(); }
      else { k.wd = nullptr; k.gd = nullptr; k.bd = nullptr; }
      L[bi++] = k;
    }
  const float* conv11 = nextf();  // (128,16,4,4)
  const float* conv21 = nextf();  // (128,8,8,8)
  const float* conv31 = nextf();  // (128,4,16,16)
  const float* conv41 = nextf();  // (128,2,32,32)
  const float* convW  = nextf();  // (128,128,3,3)
  const float* bng    = nextf();
  const float* bnb    = nextf();
  const float* hw_[4]; const float* hb_[4];
  for (int i = 0; i < 4; ++i) { hw_[i] = nextf(); hb_[i] = nextf(); }
  const float* fcw = nextf();
  const float* fcb = nextf();

  // workspace layout (floats)
  size_t off = 0;
  auto wsf = [&](size_t n) { float* p = (float*)d_ws + off; off += n; return p; };
  float* meanb   = wsf(512);
  float* varb    = wsf(512);
  float* bufStem = wsf((size_t)16 * 64 * 112 * 112);
  float* bufA    = wsf((size_t)16 * 64 * 56 * 56);
  float* bufB    = wsf((size_t)16 * 64 * 56 * 56);
  float* tmpA    = wsf((size_t)16 * 64 * 56 * 56);
  float* tmpSC   = wsf((size_t)16 * 64 * 56 * 56);
  float* x1      = wsf((size_t)16 * 64 * 56 * 56);
  float* x2      = wsf((size_t)16 * 128 * 28 * 28);
  float* x3      = wsf((size_t)16 * 256 * 14 * 14);
  float* x4      = wsf((size_t)16 * 512 * 7 * 7);
  float* code1   = wsf(16 * 64 * 4);
  float* code2   = wsf(16 * 128 * 4);
  float* code3   = wsf(16 * 256 * 4);
  float* code4   = wsf(16 * 512 * 4);
  float* a1      = wsf(16 * 128 * 11 * 11);
  float* a2      = wsf(16 * 128 * 11 * 11);
  float* a3      = wsf(16 * 128 * 11 * 11);
  float* a4      = wsf(16 * 128 * 11 * 11);
  float* xcat    = wsf(16 * 128 * 11 * 11);
  float* xcb     = wsf(16 * 128 * 11 * 11);
  float* headtmp = wsf(16 * 4 * 11 * 11);
  float* xmeanb  = wsf(16 * 121);
  float* gmeanb  = wsf(121);

  const int B = 16;

  auto run_conv = [&](const float* in, const float* w, float* out,
                      int Bn, int Cin, int H, int W, int Cout,
                      int KH, int KW, int s, int p) {
    int Ho = (H + 2 * p - KH) / s + 1;
    int Wo = (W + 2 * p - KW) / s + 1;
    long m32    = (Cout + 31) / 32;
    long ntiles = ((long)Bn * Ho * Wo + 15) / 16;
    long ngroup = (ntiles + 3) / 4;                 // 4 waves (N-tiles) per block
    int blocks  = (int)(m32 * ngroup);
    conv_wmma_kernel<<<blocks, 128, 0, stream>>>(in, w, out, Bn, Cin, H, W,
                                                 Cout, Ho, Wo, KH, KW, s, p);
  };
  auto run_bn = [&](float* x, const float* g, const float* b, const float* res,
                    int Bn, int C, int HW, int relu) {
    bn_stats_kernel<<<C, 256, 0, stream>>>(x, meanb, varb, Bn, C, HW);
    size_t total = (size_t)Bn * C * HW;
    bn_apply_kernel<<<(int)((total + 255) / 256), 256, 0, stream>>>(
        x, meanb, varb, g, b, res, x, C, HW, total, relu);
  };
  auto run_block = [&](const float* inp, float* outp, const Blk& k,
                       int Cin, int H, int W, int Cout, int stride) {
    int Ho = (H + 2 - 3) / stride + 1;
    int Wo = (W + 2 - 3) / stride + 1;
    run_conv(inp, k.w1, tmpA, B, Cin, H, W, Cout, 3, 3, stride, 1);
    run_bn(tmpA, k.g1, k.b1, nullptr, B, Cout, Ho * Wo, 1);
    run_conv(tmpA, k.w2, outp, B, Cout, Ho, Wo, Cout, 3, 3, 1, 1);
    const float* sc = inp;
    if (k.wd) {
      run_conv(inp, k.wd, tmpSC, B, Cin, H, W, Cout, 1, 1, stride, 0);
      run_bn(tmpSC, k.gd, k.bd, nullptr, B, Cout, Ho * Wo, 0);
      sc = tmpSC;
    }
    run_bn(outp, k.g2, k.b2, sc, B, Cout, Ho * Wo, 1);
  };

  // stem: conv7x7 s2 p3 -> bn -> relu -> maxpool3s2
  run_conv(X, stem_w, bufStem, B, 3, 224, 224, 64, 7, 7, 2, 3);
  run_bn(bufStem, stem_g, stem_b, nullptr, B, 64, 112 * 112, 1);
  {
    int total = B * 64 * 56 * 56;
    maxpool3s2_kernel<<<(total + 255) / 256, 256, 0, stream>>>(
        bufStem, bufA, B * 64, 112, 112, 56, 56);
  }

  run_block(bufA, bufB, L[0], 64, 56, 56, 64, 1);
  run_block(bufB, x1,   L[1], 64, 56, 56, 64, 1);
  run_block(x1, bufA,   L[2], 64, 56, 56, 128, 2);
  run_block(bufA, x2,   L[3], 128, 28, 28, 128, 1);
  run_block(x2, bufA,   L[4], 128, 28, 28, 256, 2);
  run_block(bufA, x3,   L[5], 256, 14, 14, 256, 1);
  run_block(x3, bufA,   L[6], 256, 14, 14, 512, 2);
  run_block(bufA, x4,   L[7], 512, 7, 7, 512, 1);

  // hash coding
  hash_coding_kernel<<<(B * 64 + 255) / 256, 256, 0, stream>>>(x1, code1, B, 64, 56, 56);
  hash_coding_kernel<<<(B * 128 + 255) / 256, 256, 0, stream>>>(x2, code2, B, 128, 28, 28);
  hash_coding_kernel<<<(B * 256 + 255) / 256, 256, 0, stream>>>(x3, code3, B, 256, 14, 14);
  hash_coding_kernel<<<(B * 512 + 255) / 256, 256, 0, stream>>>(x4, code4, B, 512, 7, 7);

  // pixel-shuffle side branches -> a1..a4, all (16,128,11,11)
  {
    size_t n = (size_t)B * 16 * 112 * 112;
    pixel_shuffle_kernel<<<(int)((n + 255) / 256), 256, 0, stream>>>(x1, tmpA, B, 16, 2, 56, 56);
    run_conv(tmpA, conv11, a1, B, 16, 112, 112, 128, 4, 4, 11, 1);
    n = (size_t)B * 8 * 112 * 112;
    pixel_shuffle_kernel<<<(int)((n + 255) / 256), 256, 0, stream>>>(x2, tmpA, B, 8, 4, 28, 28);
    run_conv(tmpA, conv21, a2, B, 8, 112, 112, 128, 8, 8, 11, 3);
    n = (size_t)B * 4 * 112 * 112;
    pixel_shuffle_kernel<<<(int)((n + 255) / 256), 256, 0, stream>>>(x3, tmpA, B, 4, 8, 14, 14);
    run_conv(tmpA, conv31, a3, B, 4, 112, 112, 128, 16, 16, 10, 2);
    n = (size_t)B * 2 * 112 * 112;
    pixel_shuffle_kernel<<<(int)((n + 255) / 256), 256, 0, stream>>>(x4, tmpA, B, 2, 16, 7, 7);
    run_conv(tmpA, conv41, a4, B, 2, 112, 112, 128, 32, 32, 8, 0);
  }

  size_t nA = (size_t)B * 128 * 11 * 11;
  add4_kernel<<<(int)((nA + 255) / 256), 256, 0, stream>>>(a4, a3, a2, a1, xcat, nA);

  // output layout (floats): out(112) cen(256) m2(256) m3(256) m1(256)
  //                         o1(7744) o2 o3 o4 xf(1936)
  float* out_f = (float*)d_out;
  float* o_out  = out_f;
  float* o_cen  = out_f + 112;
  float* o_m2   = out_f + 368;
  float* o_m3   = out_f + 624;
  float* o_m1   = out_f + 880;
  float* o_head[4] = { out_f + 1136, out_f + 8880, out_f + 16624, out_f + 24368 };
  float* o_xf   = out_f + 32112;

  // heads: conv(sigmoid(a_i)) + bias -> flat
  const float* aArr[4] = { a1, a2, a3, a4 };
  for (int i = 0; i < 4; ++i) {
    sigmoid_kernel<<<(int)((nA + 255) / 256), 256, 0, stream>>>(aArr[i], tmpA, nA);
    run_conv(tmpA, hw_[i], headtmp, B, 128, 11, 11, 4, 3, 3, 1, 1);
    size_t nH = (size_t)B * 4 * 121;
    bias_add_kernel<<<(int)((nH + 255) / 256), 256, 0, stream>>>(
        headtmp, hb_[i], o_head[i], 4, 121, nH);
  }

  // clustering
  kmeans4_kernel<<<B, 64, 0, stream>>>(code4, o_cen, 512);
  cluster_means_kernel<<<B, 64, 0, stream>>>(code2, o_cen, o_m2, 128);
  cluster_means_kernel<<<B, 64, 0, stream>>>(code3, o_cen, o_m3, 256);
  cluster_means_kernel<<<B, 64, 0, stream>>>(code1, o_cen, o_m1, 64);

  // xc = bn(conv(x_cat)) ; xf = mean_c + mean_bc ; out = xf @ fc_w.T + fc_b
  run_conv(xcat, convW, xcb, B, 128, 11, 11, 128, 3, 3, 1, 1);
  run_bn(xcb, bng, bnb, nullptr, B, 128, 121, 0);
  mean_c_kernel<<<(B * 121 + 255) / 256, 256, 0, stream>>>(xcb, xmeanb, B, 128, 121);
  mean_bc_kernel<<<1, 128, 0, stream>>>(xcb, gmeanb, B, 128, 121);
  xf_kernel<<<(B * 121 + 255) / 256, 256, 0, stream>>>(xmeanb, gmeanb, o_xf, B, 121);
  fc_kernel<<<1, 128, 0, stream>>>(o_xf, fcw, fcb, o_out, B, 121, 7);
}
